// MultiHeadAttentionBlock_6648609374677
// MI455X (gfx1250) — matrix operand order repaired and checked by reading, not compiled, not run on hardware
//
#include <hip/hip_runtime.h>
#include <hip/hip_bf16.h>
#include <stdint.h>

#define BB  8
#define NN  1024
#define DD  1024
#define HH  16
#define DHH 64

typedef __bf16 bf16_t;
typedef __attribute__((ext_vector_type(16))) __bf16 v16bf;
typedef __attribute__((ext_vector_type(8)))  __bf16 v8bf;
typedef __attribute__((ext_vector_type(8)))  float  v8f;

static __device__ __forceinline__ v8f vzero8() {
  v8f z = {0.f, 0.f, 0.f, 0.f, 0.f, 0.f, 0.f, 0.f};
  return z;
}

static __device__ __forceinline__ v16bf cat8(v8bf a, v8bf b) {
  return __builtin_shufflevector(a, b, 0,1,2,3,4,5,6,7,8,9,10,11,12,13,14,15);
}

// D = A*B + C, bf16 inputs, f32 accum (codegen-confirmed builtin)
static __device__ __forceinline__ v8f wmma_bf16(v16bf a, v16bf b, v8f c) {
  return __builtin_amdgcn_wmma_f32_16x16x32_bf16(false, a, false, b, (short)0, c,
                                                 false, false);
}

// A-matrix 16x32 fragment: lane row = lane&15, K = hi*8+{0..7}, hi*8+16+{0..7}
static __device__ __forceinline__ v16bf ld_frag_a(const bf16_t* base, int ld, int lane) {
  const int row = lane & 15, hi = lane >> 4;
  const bf16_t* p = base + (size_t)row * ld + hi * 8;
  v8bf lo  = *(const v8bf*)p;
  v8bf hiv = *(const v8bf*)(p + 16);
  return cat8(lo, hiv);
}

// B-matrix 32x16 fragment from "N-major" storage (column's K-vector contiguous):
// lane col = lane&15, K = hi*8+{0..7}, hi*8+16+{0..7}
static __device__ __forceinline__ v16bf ld_frag_b(const bf16_t* base, int ld, int lane) {
  const int col = lane & 15, hi = lane >> 4;
  const bf16_t* p = base + (size_t)col * ld + hi * 8;
  v8bf lo  = *(const v8bf*)p;
  v8bf hiv = *(const v8bf*)(p + 16);
  return cat8(lo, hiv);
}

// async-DMA a [128 rows x 32 k] bf16 tile into LDS (stride 40 elems = 80B).
// Uniform trip count: no exec-mask loop around the async issue.
static __device__ __forceinline__ void stage_a_async(uint32_t lb, uint64_t sbase,
                                                     int tid, int rowbase, int k0) {
#pragma unroll
  for (int it = 0; it < 4; ++it) {
    int c = tid + it * 128;
    int row = c >> 2, ch = c & 3;
    uint32_t ldsoff = lb + (uint32_t)(row * 80 + ch * 16);
    uint32_t goff = ((uint32_t)(rowbase + row) * 1024u +
                     (uint32_t)(k0 + ch * 8)) * 2u;
    asm volatile("global_load_async_to_lds_b128 %0, %1, %2"
                 :: "v"(ldsoff), "v"(goff), "s"(sbase) : "memory");
  }
}

// ---------------------------------------------------------------- converts
__global__ void cvt_kernel(const float* __restrict__ in, bf16_t* __restrict__ out, int n) {
  int i = blockIdx.x * blockDim.x + threadIdx.x;
  int stride = gridDim.x * blockDim.x;
  for (; i < n; i += stride) out[i] = (bf16_t)in[i];
}

// transpose + convert four DxD weights: T[n][k] = (bf16)W[k][n]
__global__ __launch_bounds__(256) void wtrans_kernel(
    const float* __restrict__ W0, const float* __restrict__ W1,
    const float* __restrict__ W2, const float* __restrict__ W3,
    bf16_t* __restrict__ T0, bf16_t* __restrict__ T1,
    bf16_t* __restrict__ T2, bf16_t* __restrict__ T3) {
  const float* W; bf16_t* T;
  switch (blockIdx.z) {
    case 0: W = W0; T = T0; break;
    case 1: W = W1; T = T1; break;
    case 2: W = W2; T = T2; break;
    default: W = W3; T = T3; break;
  }
  __shared__ float tile[32][33];
  const int tx = threadIdx.x & 31, ty = threadIdx.x >> 5;
  const int nb = blockIdx.x * 32, kb = blockIdx.y * 32;
#pragma unroll
  for (int i = 0; i < 4; ++i) {
    int kr = ty + i * 8;
    tile[kr][tx] = W[(size_t)(kb + kr) * DD + nb + tx];
  }
  __syncthreads();
#pragma unroll
  for (int i = 0; i < 4; ++i) {
    int nr = ty + i * 8;
    T[(size_t)(nb + nr) * DD + kb + tx] = (bf16_t)tile[tx][nr];
  }
}

// ---------------------------------------------------------------- QKV GEMM
// C[M=8192, N=1024] = A[M,K=1024] @ WT[N,K]^T + bias -> bf16
// block 128 thr = 4 waves, tile 128x64, wave = 32 rows x 64 cols.
// Double-buffered async-to-LDS A staging overlapped with WMMA.
__global__ __launch_bounds__(128) void gemm_qkv_kernel(
    const bf16_t* __restrict__ Xb, const bf16_t* __restrict__ Yb,
    const bf16_t* __restrict__ WqT, const bf16_t* __restrict__ WkT,
    const bf16_t* __restrict__ WvT,
    const float* __restrict__ bq, const float* __restrict__ bk,
    const float* __restrict__ bv,
    bf16_t* __restrict__ Qo, bf16_t* __restrict__ Ko, bf16_t* __restrict__ Vo) {
  const int z = blockIdx.z;
  const bf16_t* A    = (z == 0) ? Xb : Yb;
  const bf16_t* WT   = (z == 0) ? WqT : (z == 1) ? WkT : WvT;
  const float*  bias = (z == 0) ? bq  : (z == 1) ? bk  : bv;
  bf16_t*       out  = (z == 0) ? Qo  : (z == 1) ? Ko  : Vo;

  __shared__ bf16_t Asm[2][128 * 40];  // double buffer, 80B row stride
  const int tid = threadIdx.x, lane = tid & 31, wave = tid >> 5;
  const int hi = lane >> 4, lo = lane & 15;
  const int rowbase = blockIdx.y * 128, colbase = blockIdx.x * 64;

  v8f acc0[4] = {vzero8(), vzero8(), vzero8(), vzero8()};
  v8f acc1[4] = {vzero8(), vzero8(), vzero8(), vzero8()};
  const uint32_t lb0 = (uint32_t)(uintptr_t)(&Asm[0][0]);
  const uint32_t lb1 = (uint32_t)(uintptr_t)(&Asm[1][0]);
  const uint64_t sbase = (uint64_t)(uintptr_t)A;

  stage_a_async(lb0, sbase, tid, rowbase, 0);
  asm volatile("s_wait_asynccnt 0" ::: "memory");
  __syncthreads();

  for (int i = 0; i < 32; ++i) {
    const int k0 = i * 32;
    const bf16_t* curA = (i & 1) ? &Asm[1][0] : &Asm[0][0];
    if (i + 1 < 32)  // prefetch next tile while WMMAs run
      stage_a_async((i & 1) ? lb0 : lb1, sbase, tid, rowbase, k0 + 32);

    v16bf af0 = ld_frag_a(curA + (wave * 32) * 40, 40, lane);
    v16bf af1 = ld_frag_a(curA + (wave * 32 + 16) * 40, 40, lane);
#pragma unroll
    for (int t = 0; t < 4; ++t) {
      v16bf bfr = ld_frag_b(WT + (size_t)(colbase + 16 * t) * DD + k0, DD, lane);
      acc0[t] = wmma_bf16(af0, bfr, acc0[t]);
      acc1[t] = wmma_bf16(af1, bfr, acc1[t]);
    }
    asm volatile("s_wait_asynccnt 0" ::: "memory");
    __syncthreads();
  }

#pragma unroll
  for (int t = 0; t < 4; ++t) {
    int col = colbase + 16 * t + lo;
    float bsv = bias[col];
#pragma unroll
    for (int r = 0; r < 8; ++r) {
      int row0 = rowbase + wave * 32 + r + 8 * hi;
      out[(size_t)row0 * DD + col]        = (bf16_t)(acc0[t][r] + bsv);
      out[(size_t)(row0 + 16) * DD + col] = (bf16_t)(acc1[t][r] + bsv);
    }
  }
}

// ---------------------------------------------------------------- attention
// flash-style: block = 4 waves, 64 query rows; wave = 16 queries x DH=64.
__global__ __launch_bounds__(128) void attn_kernel(
    const bf16_t* __restrict__ Q, const bf16_t* __restrict__ K,
    const bf16_t* __restrict__ V, float* __restrict__ O) {
  __shared__ bf16_t Kt[32 * 72];      // [key][dim], 144B row stride
  __shared__ bf16_t Vt[64 * 40];      // [dim][key], 80B row stride
  __shared__ bf16_t Pb[4][16 * 40];   // per-wave P [row][key]

  const int tid = threadIdx.x, lane = tid & 31, wave = tid >> 5;
  const int hi = lane >> 4, lo = lane & 15;
  const int bh = blockIdx.y, b = bh / HH, h = bh % HH;
  const int q0 = blockIdx.x * 64 + wave * 16;

  const bf16_t* Qb = Q + ((size_t)b * NN + q0) * DD + h * DHH;
  const bf16_t* Kb = K + (size_t)b * NN * DD + h * DHH;
  const bf16_t* Vb = V + (size_t)b * NN * DD + h * DHH;
  const uint32_t ktb = (uint32_t)(uintptr_t)(&Kt[0]);
  const uint64_t ksrc = (uint64_t)(uintptr_t)Kb;

  v16bf aq0 = ld_frag_a(Qb, DD, lane);       // dims 0..31
  v16bf aq1 = ld_frag_a(Qb + 32, DD, lane);  // dims 32..63

  float m[8], l[8];
  v8f o0 = vzero8(), o1 = vzero8(), o2 = vzero8(), o3 = vzero8();
#pragma unroll
  for (int r = 0; r < 8; ++r) { m[r] = -1e30f; l[r] = 0.f; }
  const float scale = 0.03125f;  // 1/sqrt(1024)

  for (int j = 0; j < NN; j += 32) {
    // async-DMA K tile [32 keys][64 dims] into LDS (uniform trip count)
#pragma unroll
    for (int it = 0; it < 2; ++it) {
      int c = tid + it * 128;
      int key = c >> 3, ch = c & 7;
      uint32_t ldsoff = ktb + (uint32_t)(key * 144 + ch * 16);
      uint32_t goff = ((uint32_t)(j + key) * 1024u + (uint32_t)(ch * 8)) * 2u;
      asm volatile("global_load_async_to_lds_b128 %0, %1, %2"
                   :: "v"(ldsoff), "v"(goff), "s"(ksrc) : "memory");
    }
    // stage V transposed [64 dims][32 keys] (needs B-layout: manual transpose)
#pragma unroll
    for (int it = 0; it < 2; ++it) {
      int c = tid + it * 128;
      int key = c >> 3, db = (c & 7) * 8;
      v8bf d = *(const v8bf*)(Vb + (size_t)(j + key) * DD + db);
#pragma unroll
      for (int i = 0; i < 8; ++i) Vt[(db + i) * 40 + key] = d[i];
    }
    asm volatile("s_wait_asynccnt 0" ::: "memory");
    __syncthreads();

    v8f s0 = vzero8(), s1 = vzero8();
    {
      v16bf b00 = ld_frag_b(&Kt[0], 72, lane);
      v16bf b01 = ld_frag_b(&Kt[32], 72, lane);
      v16bf b10 = ld_frag_b(&Kt[16 * 72], 72, lane);
      v16bf b11 = ld_frag_b(&Kt[16 * 72 + 32], 72, lane);
      s0 = wmma_bf16(aq0, b00, s0);
      s0 = wmma_bf16(aq1, b01, s0);
      s1 = wmma_bf16(aq0, b10, s1);
      s1 = wmma_bf16(aq1, b11, s1);
    }

#pragma unroll
    for (int r = 0; r < 8; ++r) {
      float a0 = s0[r] * scale, a1 = s1[r] * scale;
      float tm = fmaxf(a0, a1);
      tm = fmaxf(tm, __shfl_xor(tm, 1, 32));
      tm = fmaxf(tm, __shfl_xor(tm, 2, 32));
      tm = fmaxf(tm, __shfl_xor(tm, 4, 32));
      tm = fmaxf(tm, __shfl_xor(tm, 8, 32));
      float mn = fmaxf(m[r], tm);
      float corr = __expf(m[r] - mn);
      m[r] = mn;
      float p0 = __expf(a0 - mn);
      float p1 = __expf(a1 - mn);
      float rs = p0 + p1;
      rs += __shfl_xor(rs, 1, 32);
      rs += __shfl_xor(rs, 2, 32);
      rs += __shfl_xor(rs, 4, 32);
      rs += __shfl_xor(rs, 8, 32);
      l[r] = l[r] * corr + rs;
      o0[r] *= corr; o1[r] *= corr; o2[r] *= corr; o3[r] *= corr;
      int prow = r + 8 * hi;
      Pb[wave][prow * 40 + lo]      = (bf16_t)p0;
      Pb[wave][prow * 40 + lo + 16] = (bf16_t)p1;
    }

    // o += P(16x32) @ V(32x64); same-wave LDS store->load is in-order
    v16bf pf  = ld_frag_a(&Pb[wave][0], 40, lane);
    v16bf vb0 = ld_frag_b(&Vt[0],        40, lane);
    v16bf vb1 = ld_frag_b(&Vt[16 * 40],  40, lane);
    v16bf vb2 = ld_frag_b(&Vt[32 * 40],  40, lane);
    v16bf vb3 = ld_frag_b(&Vt[48 * 40],  40, lane);
    o0 = wmma_bf16(pf, vb0, o0);
    o1 = wmma_bf16(pf, vb1, o1);
    o2 = wmma_bf16(pf, vb2, o2);
    o3 = wmma_bf16(pf, vb3, o3);
    __syncthreads();
  }

#pragma unroll
  for (int r = 0; r < 8; ++r) {
    float inv = 1.f / l[r];
    int row = q0 + r + 8 * hi;
    size_t base = ((size_t)b * NN + row) * DD + h * DHH + lo;
    O[base]      = o0[r] * inv;
    O[base + 16] = o1[r] * inv;
    O[base + 32] = o2[r] * inv;
    O[base + 48] = o3[r] * inv;
  }
}

// ---------------------------------------------------------------- FFN GEMM
// out2 = res + relu(A @ WT^T + bias), f32 output. Same tiling as QKV GEMM.
__global__ __launch_bounds__(128) void gemm_ffn_kernel(
    const bf16_t* __restrict__ A, const bf16_t* __restrict__ WT,
    const float* __restrict__ bias, const float* __restrict__ res,
    float* __restrict__ out) {
  __shared__ bf16_t Asm[2][128 * 40];
  const int tid = threadIdx.x, lane = tid & 31, wave = tid >> 5;
  const int hi = lane >> 4, lo = lane & 15;
  const int rowbase = blockIdx.y * 128, colbase = blockIdx.x * 64;

  v8f acc0[4] = {vzero8(), vzero8(), vzero8(), vzero8()};
  v8f acc1[4] = {vzero8(), vzero8(), vzero8(), vzero8()};
  const uint32_t lb0 = (uint32_t)(uintptr_t)(&Asm[0][0]);
  const uint32_t lb1 = (uint32_t)(uintptr_t)(&Asm[1][0]);
  const uint64_t sbase = (uint64_t)(uintptr_t)A;

  stage_a_async(lb0, sbase, tid, rowbase, 0);
  asm volatile("s_wait_asynccnt 0" ::: "memory");
  __syncthreads();

  for (int i = 0; i < 32; ++i) {
    const int k0 = i * 32;
    const bf16_t* curA = (i & 1) ? &Asm[1][0] : &Asm[0][0];
    if (i + 1 < 32)
      stage_a_async((i & 1) ? lb0 : lb1, sbase, tid, rowbase, k0 + 32);

    v16bf af0 = ld_frag_a(curA + (wave * 32) * 40, 40, lane);
    v16bf af1 = ld_frag_a(curA + (wave * 32 + 16) * 40, 40, lane);
#pragma unroll
    for (int t = 0; t < 4; ++t) {
      v16bf bfr = ld_frag_b(WT + (size_t)(colbase + 16 * t) * DD + k0, DD, lane);
      acc0[t] = wmma_bf16(af0, bfr, acc0[t]);
      acc1[t] = wmma_bf16(af1, bfr, acc1[t]);
    }
    asm volatile("s_wait_asynccnt 0" ::: "memory");
    __syncthreads();
  }

#pragma unroll
  for (int t = 0; t < 4; ++t) {
    int col = colbase + 16 * t + lo;
    float bsv = bias[col];
#pragma unroll
    for (int r = 0; r < 8; ++r) {
      int row0 = rowbase + wave * 32 + r + 8 * hi;
      size_t i0 = (size_t)row0 * DD + col;
      size_t i1 = (size_t)(row0 + 16) * DD + col;
      float v0 = fmaxf(acc0[t][r] + bsv, 0.f);
      float v1 = fmaxf(acc1[t][r] + bsv, 0.f);
      out[i0] = res[i0] + v0;
      out[i1] = res[i1] + v1;
    }
  }
}

// ---------------------------------------------------------------- layernorm
// out = LN(a [+ res]); optional bf16 copy of result
__global__ __launch_bounds__(256) void ln_kernel(
    const float* __restrict__ a, const float* __restrict__ res,
    const float* __restrict__ g, const float* __restrict__ bb,
    float* __restrict__ out_f, bf16_t* __restrict__ out_b) {
  const int row = blockIdx.x, tid = threadIdx.x;
  const size_t base = (size_t)row * DD;
  float v[4];
#pragma unroll
  for (int i = 0; i < 4; ++i) {
    int idx = tid + i * 256;
    float t = a[base + idx];
    if (res) t += res[base + idx];
    v[i] = t;
  }
  float s = v[0] + v[1] + v[2] + v[3];
  for (int mk = 1; mk < 32; mk <<= 1) s += __shfl_xor(s, mk, 32);
  __shared__ float r1[8], r2[8];
  if ((tid & 31) == 0) r1[tid >> 5] = s;
  __syncthreads();
  float mean = 0.f;
#pragma unroll
  for (int wv = 0; wv < 8; ++wv) mean += r1[wv];
  mean *= (1.0f / DD);

  float d = 0.f;
#pragma unroll
  for (int i = 0; i < 4; ++i) { float t = v[i] - mean; d += t * t; }
  for (int mk = 1; mk < 32; mk <<= 1) d += __shfl_xor(d, mk, 32);
  if ((tid & 31) == 0) r2[tid >> 5] = d;
  __syncthreads();
  float var = 0.f;
#pragma unroll
  for (int wv = 0; wv < 8; ++wv) var += r2[wv];
  var *= (1.0f / DD);
  float rstd = rsqrtf(var + 1e-6f);

#pragma unroll
  for (int i = 0; i < 4; ++i) {
    int idx = tid + i * 256;
    float o = (v[i] - mean) * rstd * g[idx] + bb[idx];
    out_f[base + idx] = o;
    if (out_b) out_b[base + idx] = (bf16_t)o;
  }
}

// ---------------------------------------------------------------- launcher
extern "C" void kernel_launch(void* const* d_in, const int* in_sizes, int n_in,
                              void* d_out, int out_size, void* d_ws, size_t ws_size,
                              hipStream_t stream) {
  const float* x  = (const float*)d_in[0];
  const float* y  = (const float*)d_in[1];
  const float* Wq = (const float*)d_in[2];
  const float* bq = (const float*)d_in[3];
  const float* Wk = (const float*)d_in[4];
  const float* bk = (const float*)d_in[5];
  const float* Wv = (const float*)d_in[6];
  const float* bv = (const float*)d_in[7];
  const float* Wf = (const float*)d_in[8];
  const float* bf = (const float*)d_in[9];
  const float* g1 = (const float*)d_in[10];
  const float* b1 = (const float*)d_in[11];
  const float* g2 = (const float*)d_in[12];
  const float* b2 = (const float*)d_in[13];
  float* out = (float*)d_out;

  char* w = (char*)d_ws;
  const size_t MB = 1024ull * 1024ull;
  bf16_t* xb  = (bf16_t*)(w + 0 * MB);     // 16 MB
  bf16_t* yb  = (bf16_t*)(w + 16 * MB);    // 16 MB
  bf16_t* WqT = (bf16_t*)(w + 32 * MB);    // 2 MB
  bf16_t* WkT = (bf16_t*)(w + 34 * MB);    // 2 MB
  bf16_t* WvT = (bf16_t*)(w + 36 * MB);    // 2 MB
  bf16_t* WfT = (bf16_t*)(w + 38 * MB);    // 2 MB
  bf16_t* qb  = (bf16_t*)(w + 40 * MB);    // 16 MB
  bf16_t* kb  = (bf16_t*)(w + 56 * MB);    // 16 MB
  bf16_t* vb  = (bf16_t*)(w + 72 * MB);    // 16 MB
  float*  ob  = (float*)(w + 88 * MB);     // 32 MB (attention output)
  float*  o1f = (float*)(w + 120 * MB);    // 32 MB (LN1 out f32)
  bf16_t* o1b = (bf16_t*)(w + 152 * MB);   // 16 MB (LN1 out bf16)
  float*  o2f = (float*)(w + 168 * MB);    // 32 MB (FFN residual out)

  const int nElem = BB * NN * DD;
  cvt_kernel<<<2048, 256, 0, stream>>>(x, xb, nElem);
  cvt_kernel<<<2048, 256, 0, stream>>>(y, yb, nElem);
  wtrans_kernel<<<dim3(32, 32, 4), 256, 0, stream>>>(Wq, Wk, Wv, Wf,
                                                     WqT, WkT, WvT, WfT);
  gemm_qkv_kernel<<<dim3(16, 64, 3), 128, 0, stream>>>(
      xb, yb, WqT, WkT, WvT, bq, bk, bv, qb, kb, vb);
  attn_kernel<<<dim3(16, 128), 128, 0, stream>>>(qb, kb, vb, ob);
  ln_kernel<<<8192, 256, 0, stream>>>(ob, x, g1, b1, o1f, o1b);
  gemm_ffn_kernel<<<dim3(16, 64), 128, 0, stream>>>(o1b, WfT, bf, o1f, o2f);
  ln_kernel<<<8192, 256, 0, stream>>>(o2f, nullptr, g2, b2, out, nullptr);
}
